// FAGCN_Net_21801253994543
// MI455X (gfx1250) — matrix operand order repaired and checked by reading, not compiled, not run on hardware
//
#include <hip/hip_runtime.h>
#include <hip/hip_bf16.h>
#include <math.h>

// ---------------------------------------------------------------------------
// FAGCN forward for MI455X (gfx1250, wave32).
//  - lin1 (N x 512 @ 512 x 64): split-bf16 WMMA (v_wmma_f32_16x16x32_bf16),
//    fp32 accumulate; memory-bound at the 102 MB fp32 read of x (~4.4us @ 23.3TB/s).
//  - edge aggregation: per-edge coef precompute + hardware fp32 global atomics
//    (h is 12.8MB -> fully L2 resident on the 192MB L2).
//  - lin2 + log_softmax fused, one wave32 per node.
// ---------------------------------------------------------------------------

typedef __attribute__((ext_vector_type(16))) __bf16 v16bf;
typedef __attribute__((ext_vector_type(8)))  float  v8f;

#define F_IN  512
#define HDIM  64
#define CDIM  40
#define LNUM  4
#define EPS_F 0.2f

static __device__ __forceinline__ __bf16 f2bf(float f) { return (__bf16)f; }

// ---- W1 -> bf16 hi/lo split -----------------------------------------------
__global__ void k_w_split(const float* __restrict__ w, __bf16* __restrict__ hi,
                          __bf16* __restrict__ lo, int n) {
  int i = blockIdx.x * blockDim.x + threadIdx.x;
  if (i >= n) return;
  float v  = w[i];
  __bf16 h = f2bf(v);
  hi[i] = h;
  lo[i] = f2bf(v - (float)h);
}

// ---- degree / dinv --------------------------------------------------------
__global__ void k_deg_init(unsigned* __restrict__ deg, int n) {
  int i = blockIdx.x * blockDim.x + threadIdx.x;
  if (i < n) deg[i] = 1u;  // self-loop
}
__global__ void k_deg_acc(const int* __restrict__ dst, unsigned* __restrict__ deg, int e) {
  int i = blockIdx.x * blockDim.x + threadIdx.x;
  if (i < e) atomicAdd(&deg[dst[i]], 1u);
}
__global__ void k_dinv(const unsigned* __restrict__ deg, float* __restrict__ dinv, int n) {
  int i = blockIdx.x * blockDim.x + threadIdx.x;
  if (i < n) dinv[i] = rsqrtf((float)deg[i]);
}

// ---- lin1: raw = relu(x @ W1^T + b1), split-bf16 WMMA ---------------------
// One wave computes a 16-row slab across all 64 output columns (4 C tiles).
__global__ __launch_bounds__(128)
void k_lin1_wmma(const float* __restrict__ x,
                 const __bf16* __restrict__ w1hi, const __bf16* __restrict__ w1lo,
                 const float* __restrict__ b1, float* __restrict__ raw, int n) {
  const int wave   = blockIdx.x * (blockDim.x >> 5) + (threadIdx.x >> 5);
  const int lane   = threadIdx.x & 31;
  const int m0     = wave * 16;
  if (m0 >= n) return;                 // uniform per wave
  const int lane15 = lane & 15;
  const int half   = lane >> 4;        // 0: lanes 0-15, 1: lanes 16-31

  v8f acc[4] = {};                     // 4 N-tiles of 16 cols, fp32 accum

  const int    row  = m0 + lane15;     // A: M = lane index
  const float* xrow = x + (size_t)row * F_IN;

  for (int k0 = 0; k0 < F_IN; k0 += 32) {
    if (k0 + 32 < F_IN) __builtin_prefetch(xrow + k0 + 32, 0, 0);

    // A fragment (16x32 bf16): lane<16 -> K = k0+0..15,
    //                          lane>=16 -> K = k0+16..31
    v16bf ahi, alo;
    const float* p0 = xrow + k0 + half * 16;
    const float* p1 = p0 + 8;
#pragma unroll
    for (int e = 0; e < 8; ++e) {
      float  v0 = p0[e];
      __bf16 h0 = f2bf(v0);
      ahi[e]     = h0;
      alo[e]     = f2bf(v0 - (float)h0);
      float  v1 = p1[e];
      __bf16 h1 = f2bf(v1);
      ahi[8 + e] = h1;
      alo[8 + e] = f2bf(v1 - (float)h1);
    }

#pragma unroll
    for (int t = 0; t < 4; ++t) {
      const int ncol = t * 16 + lane15;
      // B fragment (32x16 bf16): lane<16 -> K = k0+0..15, lane>=16 -> K = k0+16..31
      // B(k, n) = W1[n*512 + k]; 16 contiguous bf16 = one 32B load.
      const v16bf bhi = *(const v16bf*)(w1hi + (size_t)ncol * F_IN + k0 + half * 16);
      const v16bf blo = *(const v16bf*)(w1lo + (size_t)ncol * F_IN + k0 + half * 16);
      // (ahi+alo)*(bhi+blo) ~= ahi*bhi + ahi*blo + alo*bhi (fp32 accumulate)
      acc[t] = __builtin_amdgcn_wmma_f32_16x16x32_bf16(false, ahi, false, bhi,
                                                       (short)0, acc[t], false, false);
      acc[t] = __builtin_amdgcn_wmma_f32_16x16x32_bf16(false, ahi, false, blo,
                                                       (short)0, acc[t], false, false);
      acc[t] = __builtin_amdgcn_wmma_f32_16x16x32_bf16(false, alo, false, bhi,
                                                       (short)0, acc[t], false, false);
    }
  }

  // D layout: VGPR r -> M = m0 + half*8 + r, N = t*16 + lane15
#pragma unroll
  for (int t = 0; t < 4; ++t) {
    const int   ncol = t * 16 + lane15;
    const float bias = b1[ncol];
#pragma unroll
    for (int r = 0; r < 8; ++r) {
      const int m = m0 + half * 8 + r;
      float v = acc[t][r] + bias;
      raw[(size_t)m * HDIM + ncol] = v > 0.f ? v : 0.f;
    }
  }
}

// ---- per-node attention scores -------------------------------------------
__global__ void k_scores(const float* __restrict__ h, const float* __restrict__ attl,
                         const float* __restrict__ attr, float* __restrict__ al,
                         float* __restrict__ ar, int n) {
  int i = blockIdx.x * blockDim.x + threadIdx.x;
  if (i >= n) return;
  const float4* hp = (const float4*)(h + (size_t)i * HDIM);
  const float4* lp = (const float4*)attl;
  const float4* rp = (const float4*)attr;
  float sl = 0.f, sr = 0.f;
#pragma unroll
  for (int q = 0; q < HDIM / 4; ++q) {
    float4 hv = hp[q], lv = lp[q], rv = rp[q];
    sl += hv.x * lv.x + hv.y * lv.y + hv.z * lv.z + hv.w * lv.w;
    sr += hv.x * rv.x + hv.y * rv.y + hv.z * rv.z + hv.w * rv.w;
  }
  al[i] = sl;
  ar[i] = sr;
}

// ---- per-edge (+ per-node self-loop) coefficients -------------------------
__global__ void k_coef(const int* __restrict__ src, const int* __restrict__ dst,
                       const float* __restrict__ al, const float* __restrict__ ar,
                       const float* __restrict__ dinv, float* __restrict__ coef,
                       int e, int n) {
  int t = blockIdx.x * blockDim.x + threadIdx.x;
  if (t < e) {
    int s = src[t], d = dst[t];
    coef[t] = tanhf(al[s] + ar[d]) * dinv[s] * dinv[d];
  } else if (t < e + n) {
    int i = t - e;
    float di = dinv[i];
    coef[t] = tanhf(al[i] + ar[i]) * di * di;
  }
}

// ---- h_next = EPS*raw + selfcoef*h_cur ------------------------------------
__global__ void k_selfinit(const float* __restrict__ raw, const float* __restrict__ hcur,
                           const float* __restrict__ selfc, float* __restrict__ hnext,
                           int total) {
  int t = blockIdx.x * blockDim.x + threadIdx.x;
  if (t >= total) return;
  int i = t >> 6;  // HDIM = 64
  hnext[t] = EPS_F * raw[t] + selfc[i] * hcur[t];
}

// ---- edge scatter: h_next[d,:] += coef[e]*h_cur[s,:] ----------------------
__global__ void k_scatter(const int* __restrict__ src, const int* __restrict__ dst,
                          const float* __restrict__ coef, const float* __restrict__ hcur,
                          float* __restrict__ hnext, int e) {
  int t = blockIdx.x * blockDim.x + threadIdx.x;
  if (t >= e * HDIM) return;
  int ed = t >> 6, j = t & 63;
  int s = src[ed], d = dst[ed];
  float v = coef[ed] * hcur[(size_t)s * HDIM + j];
  unsafeAtomicAdd(&hnext[(size_t)d * HDIM + j], v);  // global_atomic_add_f32
}

// ---- lin2 + log_softmax fused, one wave32 per node ------------------------
__global__ __launch_bounds__(256)
void k_lin2_lsm(const float* __restrict__ h, const float* __restrict__ W2,
                const float* __restrict__ b2, float* __restrict__ out, int n) {
  int wave = blockIdx.x * (blockDim.x >> 5) + (threadIdx.x >> 5);
  int lane = threadIdx.x & 31;
  if (wave >= n) return;
  const float* hr = h + (size_t)wave * HDIM;
  const int c0 = lane, c1 = lane + 32;
  float v0 = 0.f, v1 = 0.f;
#pragma unroll 8
  for (int k = 0; k < HDIM; ++k) {
    float hv = hr[k];                       // broadcast across lanes
    v0 += hv * W2[(size_t)c0 * HDIM + k];
    if (c1 < CDIM) v1 += hv * W2[(size_t)c1 * HDIM + k];
  }
  v0 += b2[c0];
  float l1 = (c1 < CDIM) ? (v1 + b2[c1]) : -INFINITY;

  float m = fmaxf(v0, l1);
#pragma unroll
  for (int off = 16; off > 0; off >>= 1) m = fmaxf(m, __shfl_xor(m, off, 32));
  float s = __expf(v0 - m) + ((c1 < CDIM) ? __expf(l1 - m) : 0.f);
#pragma unroll
  for (int off = 16; off > 0; off >>= 1) s += __shfl_xor(s, off, 32);
  float lse = m + __logf(s);

  out[(size_t)wave * CDIM + c0] = v0 - lse;
  if (c1 < CDIM) out[(size_t)wave * CDIM + c1] = l1 - lse;
}

// ---------------------------------------------------------------------------
extern "C" void kernel_launch(void* const* d_in, const int* in_sizes, int n_in,
                              void* d_out, int out_size, void* d_ws, size_t ws_size,
                              hipStream_t stream) {
  const float* x     = (const float*)d_in[0];
  const int*   eidx  = (const int*)  d_in[1];
  const float* W1    = (const float*)d_in[2];
  const float* b1    = (const float*)d_in[3];
  const float* W2    = (const float*)d_in[4];
  const float* b2    = (const float*)d_in[5];
  const float* att_l = (const float*)d_in[6];
  const float* att_r = (const float*)d_in[7];

  const int N = in_sizes[0] / F_IN;
  const int E = in_sizes[1] / 2;
  const int* src = eidx;
  const int* dst = eidx + E;

  // workspace carve-up (256B aligned)
  char*  base = (char*)d_ws;
  size_t off  = 0;
  auto carve = [&](size_t bytes) -> char* {
    char* p = base + off;
    off = (off + bytes + 255) & ~(size_t)255;
    return p;
  };
  unsigned* deg   = (unsigned*)carve((size_t)N * 4);
  float*    dinv  = (float*)   carve((size_t)N * 4);
  float*    raw   = (float*)   carve((size_t)N * HDIM * 4);
  float*    hA    = (float*)   carve((size_t)N * HDIM * 4);
  float*    hB    = (float*)   carve((size_t)N * HDIM * 4);
  float*    al    = (float*)   carve((size_t)N * 4);
  float*    ar    = (float*)   carve((size_t)N * 4);
  float*    coef  = (float*)   carve((size_t)(E + N) * 4);
  __bf16*   w1hi  = (__bf16*)  carve((size_t)HDIM * F_IN * 2);
  __bf16*   w1lo  = (__bf16*)  carve((size_t)HDIM * F_IN * 2);
  (void)ws_size; (void)n_in; (void)out_size;

  const int B = 256;
  auto gs = [](long long total, int blk) { return (int)((total + blk - 1) / blk); };

  // weight split + norm
  k_w_split <<<gs(HDIM * F_IN, B), B, 0, stream>>>(W1, w1hi, w1lo, HDIM * F_IN);
  k_deg_init<<<gs(N, B), B, 0, stream>>>(deg, N);
  k_deg_acc <<<gs(E, B), B, 0, stream>>>(dst, deg, E);
  k_dinv    <<<gs(N, B), B, 0, stream>>>(deg, dinv, N);

  // lin1: 16-row slab per wave, 4 waves per 128-thread block
  {
    int waves  = (N + 15) / 16;
    int blocks = (waves + 3) / 4;
    k_lin1_wmma<<<blocks, 128, 0, stream>>>(x, w1hi, w1lo, b1, raw, N);
  }

  // L=4 FAConv layers: raw -> hA -> hB -> hA -> hB
  const float* hcur = raw;
  float* bufs[2] = {hA, hB};
  for (int l = 0; l < LNUM; ++l) {
    float* hnext = bufs[l & 1];
    k_scores  <<<gs(N, B), B, 0, stream>>>(hcur, att_l + (size_t)l * HDIM,
                                           att_r + (size_t)l * HDIM, al, ar, N);
    k_coef    <<<gs((long long)E + N, B), B, 0, stream>>>(src, dst, al, ar, dinv,
                                                          coef, E, N);
    k_selfinit<<<gs((long long)N * HDIM, B), B, 0, stream>>>(raw, hcur, coef + E,
                                                             hnext, N * HDIM);
    k_scatter <<<gs((long long)E * HDIM, B), B, 0, stream>>>(src, dst, coef, hcur,
                                                             hnext, E);
    hcur = hnext;
  }

  // lin2 + log_softmax: one wave per node (8 waves / 256-thread block)
  {
    int blocks = (N + 7) / 8;
    k_lin2_lsm<<<blocks, 256, 0, stream>>>(hcur, W2, b2, (float*)d_out, N);
  }
}
